// GRU_2602750181500
// MI455X (gfx1250) — compile-verified, matrix operand order checked by reading, not run on hardware
//
#include <hip/hip_runtime.h>
#include <hip/hip_bf16.h>

// ---------------------------------------------------------------------------
// Bidirectional GRU for MI455X (gfx1250), wave32 + WMMA bf16.
//  Phase 0 (prep):   fp32->bf16 conversion of x / w_ih / w_hh, h0 seed, barrier cnt
//  Phase 1 (gi):     gi[dir,t,b,g] = x[tx,b,:] @ w_ih^T + b_ih   (big WMMA GEMM)
//  Phase 2 (scan):   persistent kernel, 16 WGs per direction, w_hh in VGPRs,
//                    per-step WMMA (h @ w_hh^T) + gate math + global barrier
// ---------------------------------------------------------------------------

#define T_ 2048
#define B_ 32
#define F_ 512
#define H_ 512
#define G_ 1536
#define NWG 16            // workgroups per direction in the scan

typedef __attribute__((ext_vector_type(16))) __bf16 bf16x16;
typedef __attribute__((ext_vector_type(8)))  float  v8f;

union frag16 { bf16x16 v; uint4 q[2]; };

__device__ __forceinline__ __bf16 f2bf(float x) {
  unsigned u = __float_as_uint(x);
  u = (u + 0x7FFFu + ((u >> 16) & 1u)) >> 16;           // round-to-nearest-even
  unsigned short s = (unsigned short)u;
  return __builtin_bit_cast(__bf16, s);
}

// Load one 16x32 bf16 WMMA fragment (A layout; B uses the mirrored pattern with
// "row" = output column, stride = K). Lane L<16: row=row0+L, K={k0..k0+7, k0+16..k0+23};
// lane L>=16: row=row0+L-16, K shifted by +8. Two contiguous 16B loads per lane.
__device__ __forceinline__ bf16x16 load_frag(const __bf16* base, int row0, int stride,
                                             int k0, int lane) {
  int r  = row0 + (lane & 15);
  int ks = k0 + ((lane >> 4) << 3);
  const uint4* p = (const uint4*)(base + (size_t)r * stride + ks);
  frag16 f;
  f.q[0] = p[0];        // 8 bf16 @ ks
  f.q[1] = p[2];        // 8 bf16 @ ks+16
  return f.v;
}

__device__ __forceinline__ v8f wmma_bf16(bf16x16 a, bf16x16 b, v8f c) {
  return __builtin_amdgcn_wmma_f32_16x16x32_bf16(false, a, false, b,
                                                 (short)0, c, false, false);
}

// ---------------------------------------------------------------------------
// Phase 0: conversions + init (grid-stride loops)
// ---------------------------------------------------------------------------
__global__ __launch_bounds__(256) void prep_kernel(
    const float* __restrict__ x,      const float* __restrict__ h0,
    const float* __restrict__ wih_f,  const float* __restrict__ wih_r,
    const float* __restrict__ whh_f,  const float* __restrict__ whh_r,
    __bf16* __restrict__ xb,          __bf16* __restrict__ wihb,
    __bf16* __restrict__ whhb,        __bf16* __restrict__ hbuf,
    unsigned* __restrict__ cnt)
{
  size_t i0 = (size_t)blockIdx.x * blockDim.x + threadIdx.x;
  size_t st = (size_t)gridDim.x * blockDim.x;

  const size_t NX = (size_t)T_ * B_ * F_;
  for (size_t i = i0; i < NX; i += st) xb[i] = f2bf(x[i]);

  const size_t NWI = (size_t)G_ * F_;
  for (size_t i = i0; i < NWI; i += st) {
    wihb[i]       = f2bf(wih_f[i]);
    wihb[NWI + i] = f2bf(wih_r[i]);
  }
  const size_t NWH = (size_t)G_ * H_;
  for (size_t i = i0; i < NWH; i += st) {
    whhb[i]       = f2bf(whh_f[i]);
    whhb[NWH + i] = f2bf(whh_r[i]);
  }
  // hbuf layout: [dir][parity][B][H]; seed parity 0 from h0 = [1][2][B][H]
  const size_t NB = (size_t)2 * B_ * H_;
  for (size_t i = i0; i < NB; i += st) {
    size_t dir = i / ((size_t)B_ * H_);
    size_t rem = i % ((size_t)B_ * H_);
    hbuf[dir * 2 * B_ * H_ + rem] = f2bf(h0[i]);
  }
  if (i0 < 2) cnt[i0] = 0u;
}

// ---------------------------------------------------------------------------
// Phase 1: gi = x @ w_ih^T + b_ih for both directions (time-reversed for dir 1)
// grid = (G/64, T*B/64, 2), block = 256 (8 waves: 2 M-subtiles x 4 N-subtiles)
// Each 32-row M-subtile is exactly one time step (B == 32).
// x (67 MB bf16) and w_ih (3 MB bf16) both live in the 192 MB L2 across re-reads.
// ---------------------------------------------------------------------------
__global__ __launch_bounds__(256) void gi_gemm_kernel(
    const __bf16* __restrict__ xb,     // [T*B][F] bf16
    const __bf16* __restrict__ wihb,   // [2][G][F] bf16
    const float*  __restrict__ bih_f,  const float* __restrict__ bih_r,
    float* __restrict__ gi)            // [2][T][B][G] f32
{
  const int dir   = blockIdx.z;
  const int lane  = threadIdx.x & 31;
  const int wave  = threadIdx.x >> 5;
  const int tstep = blockIdx.y * 2 + (wave >> 2);          // scan-time index
  const int tx    = dir ? (T_ - 1 - tstep) : tstep;        // source time in x
  const int gcol0 = blockIdx.x * 64 + (wave & 3) * 16;     // this wave's 16 gate cols

  const __bf16* wb   = wihb + (size_t)dir * G_ * F_;
  const __bf16* xrow = xb + (size_t)tx * B_ * F_;

  v8f acc0 = {}; v8f acc1 = {};
#pragma unroll
  for (int kk = 0; kk < F_ / 32; ++kk) {
    const int k0 = kk * 32;
    bf16x16 bf = load_frag(wb,   gcol0, F_, k0, lane);
    bf16x16 a0 = load_frag(xrow, 0,     F_, k0, lane);
    bf16x16 a1 = load_frag(xrow, 16,    F_, k0, lane);
    acc0 = wmma_bf16(a0, bf, acc0);
    acc1 = wmma_bf16(a1, bf, acc1);
  }

  const float* bih = dir ? bih_r : bih_f;
  const int n     = gcol0 + (lane & 15);
  const float bias = bih[n];
  const int rbase = (lane >> 4) * 8;                       // C/D layout M offset
  float* grow = gi + (((size_t)dir * T_ + tstep) * B_) * G_ + n;
#pragma unroll
  for (int r = 0; r < 8; ++r) {
    const int b0 = rbase + r;
    grow[(size_t)b0 * G_]        = acc0[r] + bias;         // batch rows 0..15
    grow[(size_t)(b0 + 16) * G_] = acc1[r] + bias;         // batch rows 16..31
  }
}

// ---------------------------------------------------------------------------
// Phase 2: sequential scan. grid = (NWG, 2 dirs), block = 192 (6 waves).
// Each WG owns 32 hidden columns (96 gate columns = 3 gates x 32).
// Each wave: 16 gate columns, w_hh fragments held in 128 VGPRs for all 2048 steps.
// Full h (bf16, 32KB) staged to LDS each step; gates exchanged via LDS;
// cross-WG step barrier via monotonic agent-scope atomic counter + s_sleep.
// ---------------------------------------------------------------------------
__global__ __launch_bounds__(192) void gru_scan_kernel(
    const __bf16* __restrict__ whhb,   // [2][G][H] bf16
    const float*  __restrict__ bhh_f,  const float* __restrict__ bhh_r,
    const float*  __restrict__ gi,     // [2][T][B][G] f32
    const float*  __restrict__ h0,     // [1][2][B][H] f32
    __bf16* __restrict__ hbuf,         // [2][2][B][H] bf16 (double-buffered)
    unsigned* __restrict__ cnt,        // [2]
    float* __restrict__ out)           // [T][B][2H] f32  ++  [2][B][H] f32
{
  __shared__ __bf16 hA[B_][H_];        // 32 KB: full previous h, bf16
  __shared__ float  gh[3][B_][32];     // 12 KB: r/z/n pre-activations (h part)
  __shared__ float  hown[B_][32];      //  4 KB: owned hidden columns, fp32

  const int dir  = blockIdx.y;
  const int wg   = blockIdx.x;
  const int tid  = threadIdx.x;
  const int lane = tid & 31;
  const int wave = tid >> 5;           // 0..5
  const int gate = wave >> 1;          // 0=r 1=z 2=n
  const int sub  = wave & 1;
  const int col0 = wg * 32;                          // owned hidden col base
  const int gcol0 = gate * H_ + col0 + sub * 16;     // wave's gate col base

  const __bf16* wb  = whhb + (size_t)dir * G_ * H_;
  const float*  bhh = dir ? bhh_r : bhh_f;
  const float*  giD = gi + (size_t)dir * T_ * B_ * G_;
  __bf16*       hb  = hbuf + (size_t)dir * 2 * B_ * H_;
  unsigned*     myc = cnt + dir;

  // Pin this wave's w_hh^T slice in VGPRs: 16 K-fragments x 32B = 128 VGPRs/lane.
  bf16x16 bfrag[16];
#pragma unroll
  for (int kk = 0; kk < 16; ++kk)
    bfrag[kk] = load_frag(wb, gcol0, H_, kk * 32, lane);
  const float bias = bhh[gcol0 + (lane & 15)];

  // init owned fp32 h from h0
  for (int i = tid; i < B_ * 32; i += 192) {
    const int b = i >> 5, j = i & 31;
    hown[b][j] = h0[((size_t)dir * B_ + b) * H_ + col0 + j];
  }
  __syncthreads();

  for (int t = 0; t < T_; ++t) {
    // stage full previous h into LDS (written by all WGs at step t-1)
    const __bf16* hsrc = hb + (size_t)(t & 1) * B_ * H_;
    for (int i = tid * 8; i < B_ * H_; i += 192 * 8)
      *(uint4*)&hA[0][i] = *(const uint4*)&hsrc[i];
    __syncthreads();

    // gh = h @ w_hh^T  (32 x 16 per wave, K = 512)
    v8f acc0 = {}; v8f acc1 = {};
#pragma unroll
    for (int kk = 0; kk < 16; ++kk) {
      bf16x16 a0 = load_frag(&hA[0][0], 0,  H_, kk * 32, lane);
      bf16x16 a1 = load_frag(&hA[0][0], 16, H_, kk * 32, lane);
      acc0 = wmma_bf16(a0, bfrag[kk], acc0);
      acc1 = wmma_bf16(a1, bfrag[kk], acc1);
    }
    {
      const int n  = sub * 16 + (lane & 15);
      const int rb = (lane >> 4) * 8;
#pragma unroll
      for (int r = 0; r < 8; ++r) {
        gh[gate][rb + r][n]      = acc0[r] + bias;
        gh[gate][rb + r + 16][n] = acc1[r] + bias;
      }
    }
    __syncthreads();

    // gate math for owned columns; publish bf16 h_new, write output row
    const float* git = giD + (size_t)t * B_ * G_;
    for (int i = tid; i < B_ * 32; i += 192) {
      const int b = i >> 5, j = i & 31;
      const int c = col0 + j;
      const float rx = git[(size_t)b * G_ + c];
      const float zx = git[(size_t)b * G_ + H_ + c];
      const float nx = git[(size_t)b * G_ + 2 * H_ + c];
      const float r  = 1.f / (1.f + __expf(-(rx + gh[0][b][j])));
      const float z  = 1.f / (1.f + __expf(-(zx + gh[1][b][j])));
      const float n  = tanhf(nx + r * gh[2][b][j]);
      const float hv = (1.f - z) * n + z * hown[b][j];
      hown[b][j] = hv;
      out[((size_t)t * B_ + b) * (2 * H_) + dir * H_ + c] = hv;
      hb[(size_t)((t + 1) & 1) * B_ * H_ + (size_t)b * H_ + c] = f2bf(hv);
    }

    // device-scope step barrier across this direction's NWG workgroups
    __threadfence();
    __syncthreads();
    if (tid == 0) {
      __hip_atomic_fetch_add(myc, 1u, __ATOMIC_RELEASE, __HIP_MEMORY_SCOPE_AGENT);
      const unsigned target = (unsigned)(t + 1) * NWG;
      while (__hip_atomic_load(myc, __ATOMIC_ACQUIRE, __HIP_MEMORY_SCOPE_AGENT) < target)
        __builtin_amdgcn_s_sleep(8);
    }
    __syncthreads();
    __threadfence();
  }

  // final hidden states: out tail = [1][2][B][H]
  for (int i = tid; i < B_ * 32; i += 192) {
    const int b = i >> 5, j = i & 31;
    out[(size_t)T_ * B_ * 2 * H_ + ((size_t)dir * B_ + b) * H_ + col0 + j] = hown[b][j];
  }
}

// ---------------------------------------------------------------------------
// Workspace layout (bytes, 256-aligned):
//   wihb  bf16 [2][G][F]        @ 0            (3,145,728)
//   whhb  bf16 [2][G][H]        @ 3,145,728    (3,145,728)
//   xb    bf16 [T*B][F]         @ 6,291,456    (67,108,864)
//   gi    f32  [2][T][B][G]     @ 73,400,320   (805,306,368)
//   hbuf  bf16 [2][2][B][H]     @ 878,706,688  (131,072)
//   cnt   u32  [2]              @ 878,837,760
// ---------------------------------------------------------------------------
extern "C" void kernel_launch(void* const* d_in, const int* in_sizes, int n_in,
                              void* d_out, int out_size, void* d_ws, size_t ws_size,
                              hipStream_t stream) {
  const float* x     = (const float*)d_in[0];
  const float* h0    = (const float*)d_in[1];
  const float* wih_f = (const float*)d_in[2];
  const float* bih_f = (const float*)d_in[3];
  const float* whh_f = (const float*)d_in[4];
  const float* bhh_f = (const float*)d_in[5];
  const float* wih_r = (const float*)d_in[6];
  const float* bih_r = (const float*)d_in[7];
  const float* whh_r = (const float*)d_in[8];
  const float* bhh_r = (const float*)d_in[9];
  float* out = (float*)d_out;

  char* ws = (char*)d_ws;
  __bf16*   wihb = (__bf16*)(ws + 0);
  __bf16*   whhb = (__bf16*)(ws + 3145728);
  __bf16*   xb   = (__bf16*)(ws + 6291456);
  float*    gi   = (float*) (ws + 73400320);
  __bf16*   hbuf = (__bf16*)(ws + 878706688ull);
  unsigned* cnt  = (unsigned*)(ws + 878837760ull);

  prep_kernel<<<2048, 256, 0, stream>>>(x, h0, wih_f, wih_r, whh_f, whh_r,
                                        xb, wihb, whhb, hbuf, cnt);

  dim3 g1(G_ / 64, (T_ * B_) / 64, 2);
  gi_gemm_kernel<<<g1, 256, 0, stream>>>(xb, wihb, bih_f, bih_r, gi);

  dim3 g2(NWG, 2);
  gru_scan_kernel<<<g2, 192, 0, stream>>>(whhb, bhh_f, bhh_r, gi, h0,
                                          hbuf, cnt, out);
}
